// AttentivePolicy_9594956939918
// MI455X (gfx1250) — matrix operand order repaired and checked by reading, not compiled, not run on hardware
//
#include <hip/hip_runtime.h>
#include <hip/hip_bf16.h>

// ---------------------------------------------------------------------------
// AttentivePolicy forward for MI455X (gfx1250, wave32, WMMA).
// Big GEMMs (keys/values/q-proj/FFN/extractors) use v_wmma_f32_16x16x32_bf16.
// Weights are pre-transposed (NxK) so B-fragments are contiguous 32B reads.
// Irregular pieces (softmax attention, layernorm, embed MLP) are VALU kernels.
// ---------------------------------------------------------------------------

typedef __attribute__((ext_vector_type(16))) __bf16 v16bf;
typedef __attribute__((ext_vector_type(8)))  __bf16 v8bf;
typedef __attribute__((ext_vector_type(8)))  float  v8f;

#define BATCH   1024
#define NTOK    256
#define DMODEL  512
#define NHEAD   8
#define DHEAD   64
#define DFF     2048
#define LN_EPS  1e-5f

// ---------------------------------------------------------------------------
// Weight prep kernels (tiny, once per call). All produce TRANSPOSED bf16
// weights: Wt[n * K + k] so the GEMM B operand is (N x K) row-major.
// ---------------------------------------------------------------------------

// src (R, C) f32 row-major -> dst (C, R) bf16 row-major
__global__ void transpose_bf16_kernel(const float* __restrict__ src,
                                      __bf16* __restrict__ dst, int R, int C) {
  int i = blockIdx.x * blockDim.x + threadIdx.x;
  if (i >= R * C) return;
  int r = i / C, c = i % C;
  dst[(size_t)c * R + r] = (__bf16)src[i];
}

// W (8,512,64) -> dst (N=512 rows h*64+k, K=512 cols d), row-major bf16
__global__ void permute_hdk_t_kernel(const float* __restrict__ w,
                                     __bf16* __restrict__ dst) {
  int i = blockIdx.x * blockDim.x + threadIdx.x;  // 8*512*64
  if (i >= NHEAD * DMODEL * DHEAD) return;
  int h = i / (DMODEL * DHEAD);
  int d = (i / DHEAD) % DMODEL;
  int k = i % DHEAD;
  dst[(size_t)(h * DHEAD + k) * DMODEL + d] = (__bf16)w[i];
}

// extract w2 (256,10) -> transposed + padded (16, 256) bf16
__global__ void pad_w2_t_kernel(const float* __restrict__ w,
                                __bf16* __restrict__ dst) {
  int i = blockIdx.x * blockDim.x + threadIdx.x;  // 16*256
  if (i >= 16 * 256) return;
  int c = i / 256, r = i % 256;
  dst[i] = (c < 10) ? (__bf16)w[r * 10 + c] : (__bf16)0.0f;
}

// extract b2 (10,) -> (16,) zero padded, f32
__global__ void pad_b2_kernel(const float* __restrict__ b, float* __restrict__ dst) {
  int i = threadIdx.x;
  if (i < 16) dst[i] = (i < 10) ? b[i] : 0.0f;
}

// ---------------------------------------------------------------------------
// Embedding: pos-MLP (2->128 relu ->256) || class embedding (256)
// One block per token, 256 threads. Writes bf16 tokens + f32 ego rows.
// ---------------------------------------------------------------------------
__global__ __launch_bounds__(256)
void embed_kernel(const float* __restrict__ M_pad,
                  const float* __restrict__ pos_w1, const float* __restrict__ pos_b1,
                  const float* __restrict__ pos_w2, const float* __restrict__ pos_b2,
                  const float* __restrict__ class_emb,
                  __bf16* __restrict__ x_bf, float* __restrict__ ego) {
  __shared__ float h[128];
  int t = blockIdx.x;                     // token index in [0, BATCH*NTOK)
  float x0 = M_pad[(size_t)t * 3 + 0];
  float x1 = M_pad[(size_t)t * 3 + 1];
  int cls = (int)M_pad[(size_t)t * 3 + 2];
  cls = max(0, min(6, cls));
  int tid = threadIdx.x;
  if (tid < 128) {
    float v = fmaf(x0, pos_w1[tid], fmaf(x1, pos_w1[128 + tid], pos_b1[tid]));
    h[tid] = fmaxf(v, 0.0f);
  }
  __syncthreads();
  float acc = pos_b2[tid];
  #pragma unroll 8
  for (int i = 0; i < 128; ++i)
    acc = fmaf(h[i], pos_w2[i * 256 + tid], acc);
  float ce = class_emb[cls * 256 + tid];
  x_bf[(size_t)t * DMODEL + tid] = (__bf16)acc;
  x_bf[(size_t)t * DMODEL + 256 + tid] = (__bf16)ce;
  if ((t & (NTOK - 1)) == 0) {            // ego token (n==0)
    int b = t >> 8;
    ego[(size_t)b * DMODEL + tid] = acc;
    ego[(size_t)b * DMODEL + 256 + tid] = ce;
  }
}

// ---------------------------------------------------------------------------
// bf16 WMMA GEMM:  C(MxN) = op(A(MxK) * B(KxN) + bias), B given as Bt (NxK).
// block = 128 threads (4 waves). Wave w computes a 32x64 tile:
//   rows m0..m0+31, cols n0 = blockIdx.y*256 + w*64.
// LDS: A tile 32x32 (row-major), B tile stored as sB[col][k] (fragment-ready)
// so every B fragment is one contiguous 32-byte LDS read.
// Requires M%32==0, K%32==0; N tile edges guarded (zero pad / masked store).
// ---------------------------------------------------------------------------
__global__ __launch_bounds__(128)
void gemm_bf16_kernel(const __bf16* __restrict__ A, int lda,
                      const __bf16* __restrict__ Bt, int ldbt,  // (N x K)
                      const float* __restrict__ bias,
                      void* __restrict__ Cout, int ldc,
                      int M, int N, int K, int relu, int out_bf16) {
  __shared__ __bf16 sA[32 * 32];
  __shared__ __bf16 sB[4][64 * 32];        // [col][k]
  const int wave = threadIdx.x >> 5;
  const int lane = threadIdx.x & 31;
  const int m0 = blockIdx.x * 32;
  const int n0 = blockIdx.y * 256 + wave * 64;

  v8f acc[2][4] = {};

  const int r = lane & 15;
  const int kb = (lane >> 4) * 16;   // A-fragment K base within tile
  const int kb2 = (lane >> 4) * 16;  // B-fragment K base within tile

  for (int k0 = 0; k0 < K; k0 += 32) {
    // ---- stage A tile (32x32 bf16 = 2KB): one uint4 (16B) per thread
    {
      int row = threadIdx.x >> 2;          // 0..31
      int col8 = (threadIdx.x & 3) * 8;    // 0,8,16,24
      const __bf16* src = A + (size_t)(m0 + row) * lda + k0 + col8;
      *(uint4*)&sA[row * 32 + col8] = *(const uint4*)src;
      if (k0 + 32 < K)
        __builtin_prefetch(src + 32, 0, 0);    // global_prefetch_b8
    }
    // ---- stage this wave's B tile: 64 cols x 32 k, 2 cols per lane (wide)
    #pragma unroll
    for (int rr = 0; rr < 2; ++rr) {
      int row = rr * 32 + lane;            // B column index within tile
      __bf16* dst = &sB[wave][row * 32];
      if (n0 + row < N) {
        const __bf16* src = Bt + (size_t)(n0 + row) * ldbt + k0;
        #pragma unroll
        for (int c = 0; c < 32; c += 8)
          *(uint4*)&dst[c] = *(const uint4*)&src[c];
        if (k0 + 32 < K) __builtin_prefetch(src + 32, 0, 0);
      } else {
        #pragma unroll
        for (int c = 0; c < 32; c += 8)
          *(uint4*)&dst[c] = make_uint4(0u, 0u, 0u, 0u);
      }
    }
    __syncthreads();

    // ---- A fragments (16x32 each): lane holds K-chunk {kb..kb+15}
    v16bf afrag[2];
    #pragma unroll
    for (int mi = 0; mi < 2; ++mi) {
      union { v16bf v; v8bf h[2]; } u;
      u.h[0] = *(const v8bf*)&sA[(mi * 16 + r) * 32 + kb];
      u.h[1] = *(const v8bf*)&sA[(mi * 16 + r) * 32 + kb + 8];
      afrag[mi] = u.v;
    }
    // ---- 4 B fragments (one 32B contiguous read each) + 8 WMMAs
    #pragma unroll
    for (int j = 0; j < 4; ++j) {
      const int c = (lane & 15) + 16 * j;
      v16bf bfrag = *(const v16bf*)&sB[wave][c * 32 + kb2];
      #pragma unroll
      for (int mi = 0; mi < 2; ++mi)
        acc[mi][j] = __builtin_amdgcn_wmma_f32_16x16x32_bf16(
            false, afrag[mi], false, bfrag, (short)0, acc[mi][j], false, false);
    }
    __syncthreads();
  }

  // ---- store: C layout — lane&15 = column, VGPR rr -> row rr + 8*(lane>>4)
  #pragma unroll
  for (int mi = 0; mi < 2; ++mi) {
    const int rbase = m0 + mi * 16 + (lane >> 4) * 8;
    #pragma unroll
    for (int j = 0; j < 4; ++j) {
      int col = n0 + 16 * j + (lane & 15);
      if (col >= N) continue;
      float bv = bias ? bias[col] : 0.0f;
      #pragma unroll
      for (int rr = 0; rr < 8; ++rr) {
        float v = acc[mi][j][rr] + bv;
        if (relu) v = fmaxf(v, 0.0f);
        size_t idx = (size_t)(rbase + rr) * ldc + col;
        if (out_bf16) ((__bf16*)Cout)[idx] = (__bf16)v;
        else          ((float*)Cout)[idx] = v;
      }
    }
  }
}

// ---------------------------------------------------------------------------
// Masked-softmax ego attention. One block per batch row, 256 threads.
// scores(h,n) = q[h]·K[n,h] / 8, masked; attn_out = softmax(scores) @ V.
// ---------------------------------------------------------------------------
__global__ __launch_bounds__(256)
void attn_kernel(const float* __restrict__ q,        // (B, 512) head-major
                 const __bf16* __restrict__ Kmat,    // (B*NTOK, 512)
                 const float* __restrict__ V,        // (B*NTOK, 64)
                 const unsigned char* __restrict__ mask,  // (B, NTOK) bool
                 float* __restrict__ attn_out) {     // (B, 512)
  __shared__ float sq[DMODEL];
  __shared__ float sc[256];
  __shared__ float red[256];
  const int b = blockIdx.x;
  const int tid = threadIdx.x;
  sq[tid]       = q[(size_t)b * DMODEL + tid];
  sq[tid + 256] = q[(size_t)b * DMODEL + tid + 256];
  __syncthreads();

  for (int h = 0; h < NHEAD; ++h) {
    float s = -INFINITY;
    if (tid < NTOK - 1) {                       // token n = 1 + tid
      const __bf16* krow = Kmat + ((size_t)b * NTOK + 1 + tid) * DMODEL + h * DHEAD;
      float acc = 0.0f;
      #pragma unroll 8
      for (int k = 0; k < DHEAD; ++k)
        acc = fmaf(sq[h * DHEAD + k], (float)krow[k], acc);
      s = mask[(size_t)b * NTOK + 1 + tid] ? acc * 0.125f : -INFINITY;
    }
    sc[tid] = s;
    red[tid] = s;
    __syncthreads();
    for (int off = 128; off > 0; off >>= 1) {
      if (tid < off) red[tid] = fmaxf(red[tid], red[tid + off]);
      __syncthreads();
    }
    float mx = red[0];
    __syncthreads();
    float e = (sc[tid] == -INFINITY) ? 0.0f : __expf(sc[tid] - mx);
    sc[tid] = e;
    red[tid] = e;
    __syncthreads();
    for (int off = 128; off > 0; off >>= 1) {
      if (tid < off) red[tid] += red[tid + off];
      __syncthreads();
    }
    float inv = 1.0f / red[0];
    __syncthreads();
    if (tid < DHEAD) {
      float acc = 0.0f;
      for (int n = 0; n < NTOK - 1; ++n)
        acc = fmaf(sc[n], V[((size_t)b * NTOK + 1 + n) * DHEAD + tid], acc);
      attn_out[(size_t)b * DMODEL + h * DHEAD + tid] = acc * inv;
    }
    __syncthreads();
  }
}

// ---------------------------------------------------------------------------
// out = LayerNorm(a + c) * g + beta ; writes f32 and bf16 (for next GEMM).
// ---------------------------------------------------------------------------
__global__ __launch_bounds__(256)
void add_ln_kernel(const float* __restrict__ a, const float* __restrict__ c,
                   const float* __restrict__ g, const float* __restrict__ beta,
                   float* __restrict__ out_f32, __bf16* __restrict__ out_bf) {
  __shared__ float red[256];
  const int b = blockIdx.x, tid = threadIdx.x;
  const size_t base = (size_t)b * DMODEL;
  float v0 = a[base + tid] + c[base + tid];
  float v1 = a[base + 256 + tid] + c[base + 256 + tid];
  red[tid] = v0 + v1;
  __syncthreads();
  for (int off = 128; off > 0; off >>= 1) {
    if (tid < off) red[tid] += red[tid + off];
    __syncthreads();
  }
  float mu = red[0] * (1.0f / 512.0f);
  __syncthreads();
  float d0 = v0 - mu, d1 = v1 - mu;
  red[tid] = d0 * d0 + d1 * d1;
  __syncthreads();
  for (int off = 128; off > 0; off >>= 1) {
    if (tid < off) red[tid] += red[tid + off];
    __syncthreads();
  }
  float rstd = rsqrtf(red[0] * (1.0f / 512.0f) + LN_EPS);
  float o0 = fmaf(d0 * rstd, g[tid], beta[tid]);
  float o1 = fmaf(d1 * rstd, g[tid + 256], beta[tid + 256]);
  out_f32[base + tid] = o0;
  out_f32[base + 256 + tid] = o1;
  out_bf[base + tid] = (__bf16)o0;
  out_bf[base + 256 + tid] = (__bf16)o1;
}

__global__ void f32_to_bf16_kernel(const float* __restrict__ src,
                                   __bf16* __restrict__ dst, int n) {
  int i = blockIdx.x * blockDim.x + threadIdx.x;
  if (i < n) dst[i] = (__bf16)src[i];
}

// select per-batch extractor head by action; outs = (3, B, 16), out = (B, 10)
__global__ void select_kernel(const float* __restrict__ outs,
                              const int* __restrict__ action,
                              float* __restrict__ out) {
  int i = blockIdx.x * blockDim.x + threadIdx.x;
  if (i >= BATCH * 10) return;
  int b = i / 10, j = i % 10;
  int a = action[b];
  a = max(0, min(2, a));
  out[i] = outs[(size_t)a * BATCH * 16 + (size_t)b * 16 + j];
}

// ---------------------------------------------------------------------------
// Host side
// ---------------------------------------------------------------------------
static inline void launch_gemm(const __bf16* A, int lda, const __bf16* Bt, int ldbt,
                               const float* bias, void* C, int ldc,
                               int M, int N, int K, int relu, int out_bf16,
                               hipStream_t stream) {
  dim3 grid(M / 32, (N + 255) / 256);
  gemm_bf16_kernel<<<grid, 128, 0, stream>>>(A, lda, Bt, ldbt, bias, C, ldc,
                                             M, N, K, relu, out_bf16);
}

extern "C" void kernel_launch(void* const* d_in, const int* in_sizes, int n_in,
                              void* d_out, int out_size, void* d_ws, size_t ws_size,
                              hipStream_t stream) {
  (void)in_sizes; (void)n_in; (void)out_size; (void)ws_size;
  // Input order: top-level dict insertion order; params jax-flattened
  // (sorted keys; lists in order).
  const float* M_pad     = (const float*)d_in[0];
  const unsigned char* M_mask = (const unsigned char*)d_in[1];
  const int*   action    = (const int*)d_in[2];
  const float* W_k       = (const float*)d_in[3];
  const float* W_v       = (const float*)d_in[4];
  const float* class_emb = (const float*)d_in[5];
  const float *ex_b1[3], *ex_b2[3], *ex_w1[3], *ex_w2[3];
  for (int e = 0; e < 3; ++e) {
    ex_b1[e] = (const float*)d_in[6 + 4 * e + 0];
    ex_b2[e] = (const float*)d_in[6 + 4 * e + 1];
    ex_w1[e] = (const float*)d_in[6 + 4 * e + 2];
    ex_w2[e] = (const float*)d_in[6 + 4 * e + 3];
  }
  const float *W_q[2], *ln1_b[2], *ln1_g[2], *ln2_b[2], *ln2_g[2];
  const float *mlp_b1[2], *mlp_b2[2], *mlp_w1[2], *mlp_w2[2];
  for (int l = 0; l < 2; ++l) {
    int base = 18 + 9 * l;
    W_q[l]    = (const float*)d_in[base + 0];
    ln1_b[l]  = (const float*)d_in[base + 1];
    ln1_g[l]  = (const float*)d_in[base + 2];
    ln2_b[l]  = (const float*)d_in[base + 3];
    ln2_g[l]  = (const float*)d_in[base + 4];
    mlp_b1[l] = (const float*)d_in[base + 5];
    mlp_b2[l] = (const float*)d_in[base + 6];
    mlp_w1[l] = (const float*)d_in[base + 7];
    mlp_w2[l] = (const float*)d_in[base + 8];
  }
  const float* pos_b1 = (const float*)d_in[36];
  const float* pos_b2 = (const float*)d_in[37];
  const float* pos_w1 = (const float*)d_in[38];
  const float* pos_w2 = (const float*)d_in[39];

  // ---- workspace bump allocator
  char* p = (char*)d_ws;
  auto alloc = [&](size_t bytes) -> char* {
    char* r = p;
    p += (bytes + 255) & ~(size_t)255;
    return r;
  };
  const size_t T = (size_t)BATCH * NTOK;
  __bf16* x_bf     = (__bf16*)alloc(T * DMODEL * 2);
  __bf16* K_bf     = (__bf16*)alloc(T * DMODEL * 2);
  float*  Vbuf     = (float*)alloc(T * DHEAD * 4);
  float*  ego      = (float*)alloc((size_t)BATCH * DMODEL * 4);
  __bf16* ego_bf   = (__bf16*)alloc((size_t)BATCH * DMODEL * 2);
  float*  qbuf     = (float*)alloc((size_t)BATCH * DMODEL * 4);
  float*  attn_out = (float*)alloc((size_t)BATCH * DMODEL * 4);
  float*  x1       = (float*)alloc((size_t)BATCH * DMODEL * 4);
  __bf16* x1_bf    = (__bf16*)alloc((size_t)BATCH * DMODEL * 2);
  __bf16* hid_bf   = (__bf16*)alloc((size_t)BATCH * DFF * 2);
  float*  mlp_out  = (float*)alloc((size_t)BATCH * DMODEL * 4);
  __bf16* exhid_bf = (__bf16*)alloc((size_t)BATCH * 256 * 2);
  float*  outs     = (float*)alloc((size_t)3 * BATCH * 16 * 4);
  __bf16* WkT      = (__bf16*)alloc(512 * 512 * 2);   // (N=512, K=512)
  __bf16* WvT      = (__bf16*)alloc(64 * 512 * 2);    // (N=64,  K=512)
  __bf16 *WqT[2], *W1T[2], *W2T[2];
  for (int l = 0; l < 2; ++l) {
    WqT[l] = (__bf16*)alloc(512 * 512 * 2);           // (512, 512)
    W1T[l] = (__bf16*)alloc(2048 * 512 * 2);          // (2048, 512)
    W2T[l] = (__bf16*)alloc(512 * 2048 * 2);          // (512, 2048)
  }
  __bf16 *Ew1T[3], *Ew2T[3];
  float* Eb2p[3];
  for (int e = 0; e < 3; ++e) {
    Ew1T[e] = (__bf16*)alloc(256 * 512 * 2);          // (256, 512)
    Ew2T[e] = (__bf16*)alloc(16 * 256 * 2);           // (16, 256)
    Eb2p[e] = (float*)alloc(16 * 4);
  }

  // ---- weight prep (transposed bf16)
  const int PB = 256;
  permute_hdk_t_kernel<<<(NHEAD * DMODEL * DHEAD + PB - 1) / PB, PB, 0, stream>>>(W_k, WkT);
  transpose_bf16_kernel<<<(512 * 64 + PB - 1) / PB, PB, 0, stream>>>(W_v, WvT, 512, 64);
  for (int l = 0; l < 2; ++l) {
    permute_hdk_t_kernel<<<(NHEAD * DMODEL * DHEAD + PB - 1) / PB, PB, 0, stream>>>(W_q[l], WqT[l]);
    transpose_bf16_kernel<<<(512 * 2048 + PB - 1) / PB, PB, 0, stream>>>(mlp_w1[l], W1T[l], 512, 2048);
    transpose_bf16_kernel<<<(2048 * 512 + PB - 1) / PB, PB, 0, stream>>>(mlp_w2[l], W2T[l], 2048, 512);
  }
  for (int e = 0; e < 3; ++e) {
    transpose_bf16_kernel<<<(512 * 256 + PB - 1) / PB, PB, 0, stream>>>(ex_w1[e], Ew1T[e], 512, 256);
    pad_w2_t_kernel<<<(16 * 256 + PB - 1) / PB, PB, 0, stream>>>(ex_w2[e], Ew2T[e]);
    pad_b2_kernel<<<1, 16, 0, stream>>>(ex_b2[e], Eb2p[e]);
  }

  // ---- embedding: tokens -> x_bf (bf16) and ego rows (f32)
  embed_kernel<<<(unsigned)T, 256, 0, stream>>>(M_pad, pos_w1, pos_b1, pos_w2,
                                                pos_b2, class_emb, x_bf, ego);

  // ---- keys / values projections (the dominant GEMMs)
  launch_gemm(x_bf, DMODEL, WkT, DMODEL, nullptr, K_bf, DMODEL,
              (int)T, DMODEL, DMODEL, 0, 1, stream);
  launch_gemm(x_bf, DMODEL, WvT, DMODEL, nullptr, Vbuf, DHEAD,
              (int)T, DHEAD, DMODEL, 0, 0, stream);

  f32_to_bf16_kernel<<<(BATCH * DMODEL + 255) / 256, 256, 0, stream>>>(
      ego, ego_bf, BATCH * DMODEL);

  // ---- transformer layers
  for (int l = 0; l < 2; ++l) {
    launch_gemm(ego_bf, DMODEL, WqT[l], DMODEL, nullptr, qbuf, DMODEL,
                BATCH, DMODEL, DMODEL, 0, 0, stream);
    attn_kernel<<<BATCH, 256, 0, stream>>>(qbuf, K_bf, Vbuf, M_mask, attn_out);
    add_ln_kernel<<<BATCH, 256, 0, stream>>>(ego, attn_out, ln1_g[l], ln1_b[l],
                                             x1, x1_bf);
    launch_gemm(x1_bf, DMODEL, W1T[l], DMODEL, mlp_b1[l], hid_bf, DFF,
                BATCH, DFF, DMODEL, 1, 1, stream);
    launch_gemm(hid_bf, DFF, W2T[l], DFF, mlp_b2[l], mlp_out, DMODEL,
                BATCH, DMODEL, DFF, 0, 0, stream);
    add_ln_kernel<<<BATCH, 256, 0, stream>>>(x1, mlp_out, ln2_g[l], ln2_b[l],
                                             ego, ego_bf);
  }

  // ---- extractor heads + action select
  for (int e = 0; e < 3; ++e) {
    launch_gemm(ego_bf, DMODEL, Ew1T[e], DMODEL, ex_b1[e], exhid_bf, 256,
                BATCH, 256, DMODEL, 1, 1, stream);
    launch_gemm(exhid_bf, 256, Ew2T[e], 256, Eb2p[e],
                outs + (size_t)e * BATCH * 16, 16,
                BATCH, 16, 256, 0, 0, stream);
  }
  select_kernel<<<(BATCH * 10 + 255) / 256, 256, 0, stream>>>(outs, action,
                                                              (float*)d_out);
}
